// LstmDecoder_13950053777937
// MI455X (gfx1250) — compile-verified, matrix operand order checked by reading, not run on hardware
//
#include <hip/hip_runtime.h>

// ---------------------------------------------------------------------------
// LSTM attention decoder for MI455X (gfx1250, wave32).
// GEMMs: V_WMMA_F32_16X16X32_BF16 with double-buffered LDS staged via the
// CDNA5 async DMA path (GLOBAL_LOAD_ASYNC_TO_LDS_B128 / ASYNCcnt).
// ---------------------------------------------------------------------------

typedef __bf16 bf16;
typedef __attribute__((ext_vector_type(16))) __bf16 v16bf;
typedef __attribute__((ext_vector_type(8)))  float  v8f;

static constexpr int Vv = 1000;   // vocab
static constexpr int Vp = 1024;   // vocab padded to multiple of 128 for GEMM
static constexpr int Hh = 512;    // hidden == emb
static constexpr int Ll = 2;
static constexpr int Ss = 64;
static constexpr int Bb = 512;
static constexpr int Tt = 64;
static constexpr float NEG_FILL = -9999.9f;

union Frag16 { v16bf v; uint4 q[2]; };

__device__ __forceinline__ float wred_sum(float v) {
#pragma unroll
  for (int o = 16; o > 0; o >>= 1) v += __shfl_xor(v, o, 32);
  return v;
}
__device__ __forceinline__ float wred_max(float v) {
#pragma unroll
  for (int o = 16; o > 0; o >>= 1) v = fmaxf(v, __shfl_xor(v, o, 32));
  return v;
}
__device__ __forceinline__ float sigm(float x) { return 1.f / (1.f + __expf(-x)); }

// async DMA: copy 16B per lane, global -> LDS, tracked by ASYNCcnt
__device__ __forceinline__ void async_copy_b128(unsigned lds_off, const void* gptr) {
  asm volatile("global_load_async_to_lds_b128 %0, %1, off"
               :: "v"(lds_off), "v"((unsigned long long)(size_t)gptr)
               : "memory");
}

// ---------------------------------------------------------------------------
// NT GEMM: C[M,N] = concat(Alo,Ahi)[M,K] * W[N,K]^T + bias[N]
// A is split at column Ksplit (Ahi may be null when Ksplit == K).
// M%128==0, N%128==0, K%32==0 by construction.
// Block = 256 threads (8 waves); block tile 128x128; wave tile 32x64.
// Double-buffered LDS: chunk i+1 DMA'd while chunk i feeds the WMMAs.
// ---------------------------------------------------------------------------
template <bool WRITE_BF16>
__global__ __launch_bounds__(256) void gemm_nt(
    const bf16* __restrict__ Alo, const bf16* __restrict__ Ahi,
    const bf16* __restrict__ W, const float* __restrict__ bias,
    void* __restrict__ Cout, int M, int N, int K, int Ksplit) {
  // +8 bf16 pad -> 80B row pitch: 16B-aligned rows, conflict-free b128 reads
  __shared__ __align__(16) bf16 lA[2][128][40];
  __shared__ __align__(16) bf16 lB[2][128][40];

  const int tid  = threadIdx.x;
  const int lane = tid & 31;
  const int wv   = tid >> 5;
  const int wm   = wv >> 1;            // 0..3 (M direction)
  const int wn   = wv & 1;             // 0..1 (N direction)
  const int m0   = blockIdx.y * 128;
  const int n0   = blockIdx.x * 128;
  const int rsel = lane & 15;
  const int klo  = (lane >> 4) << 3;   // 0 or 8 (ISA 16-bit A-frag K packing)
  const int Khi  = K - Ksplit;

  // per-thread staging coordinates (fixed across chunks):
  // u=0/1 -> two consecutive 8-element (16B) pieces of one 32-wide K row
  const int sr = tid >> 1;                 // tile row 0..127
  const int sc0 = (tid & 1) * 16;          // 0 or 16

  // stage one 128x32 K-chunk of A and W into LDS buffer `buf` (4 async ops)
  auto stage = [&](int kc, int buf) {
#pragma unroll
    for (int u = 0; u < 2; ++u) {
      const int c  = sc0 + u * 8;
      const int gk = kc + c;
      const bf16* sa = (gk < Ksplit)
          ? Alo + (size_t)(m0 + sr) * Ksplit + gk
          : Ahi + (size_t)(m0 + sr) * Khi + (gk - Ksplit);
      const bf16* sw = W + (size_t)(n0 + sr) * K + gk;
      async_copy_b128((unsigned)(size_t)&lA[buf][sr][c], sa);
      async_copy_b128((unsigned)(size_t)&lB[buf][sr][c], sw);
      if (gk + 64 < K) __builtin_prefetch(sw + 64, 0, 1);  // global_prefetch_b8
    }
  };

  v8f acc[2][4] = {};
  const int nch = K >> 5;

  stage(0, 0);
  for (int ic = 0; ic < nch; ++ic) {
    const int buf = ic & 1;
    if (ic + 1 < nch) {
      stage((ic + 1) * 32, buf ^ 1);                 // overlap next chunk DMA
      asm volatile("s_wait_asynccnt 0x4" ::: "memory");  // current chunk done
    } else {
      asm volatile("s_wait_asynccnt 0x0" ::: "memory");
    }
    __syncthreads();   // all waves' DMA for `buf` visible

    // fragment loads (ds_load_b128 x2 per fragment) + 8 WMMAs
    Frag16 fa[2], fb[4];
#pragma unroll
    for (int i = 0; i < 2; ++i) {
      const int r = wm * 32 + i * 16 + rsel;
      fa[i].q[0] = *(const uint4*)&lA[buf][r][klo];
      fa[i].q[1] = *(const uint4*)&lA[buf][r][16 + klo];
    }
#pragma unroll
    for (int j = 0; j < 4; ++j) {
      const int r = wn * 64 + j * 16 + rsel;
      fb[j].q[0] = *(const uint4*)&lB[buf][r][klo];
      fb[j].q[1] = *(const uint4*)&lB[buf][r][16 + klo];
    }
#pragma unroll
    for (int i = 0; i < 2; ++i)
#pragma unroll
      for (int j = 0; j < 4; ++j)
        acc[i][j] = __builtin_amdgcn_wmma_f32_16x16x32_bf16(
            false, fa[i].v, false, fb[j].v, (short)0, acc[i][j], false, false);

    __syncthreads();   // all waves done reading `buf` before it is re-DMA'd
  }

  // epilogue: C layout -> lane gives N (lane&15), VGPR v gives M (+8 for hi half)
#pragma unroll
  for (int i = 0; i < 2; ++i) {
#pragma unroll
    for (int j = 0; j < 4; ++j) {
      const int n  = n0 + wn * 64 + j * 16 + rsel;
      const float bv = bias ? bias[n] : 0.f;
      const int mb = m0 + wm * 32 + i * 16 + ((lane >> 4) << 3);
#pragma unroll
      for (int v = 0; v < 8; ++v) {
        const float val = acc[i][j][v] + bv;
        const size_t ci = (size_t)(mb + v) * N + n;
        if constexpr (WRITE_BF16) ((bf16*)Cout)[ci] = (bf16)val;
        else                      ((float*)Cout)[ci] = val;
      }
    }
  }
}

// ------------------------- elementwise / setup kernels ----------------------

__global__ void conv_emb_kernel(bf16* __restrict__ out, const float* __restrict__ in) {
  const int idx = blockIdx.x * blockDim.x + threadIdx.x;      // Vp*H
  if (idx >= Vp * Hh) return;
  const int r = idx >> 9, e = idx & 511;
  out[idx] = (r < Vv) ? (bf16)in[(size_t)r * Hh + e] : (bf16)0.f;
}

__global__ void pack_w_kernel(bf16* __restrict__ Wl, const float* __restrict__ w_ih,
                              const float* __restrict__ w_hh) {
  const int idx = blockIdx.x * blockDim.x + threadIdx.x;      // L*4H*(2H)
  if (idx >= Ll * 4 * Hh * 2 * Hh) return;
  const int l = idx / (4 * Hh * 2 * Hh);
  const int rem = idx - l * (4 * Hh * 2 * Hh);
  const int n = rem / (2 * Hh);
  const int k = rem - n * (2 * Hh);
  const float v = (k < Hh) ? w_ih[((size_t)l * 4 * Hh + n) * Hh + k]
                           : w_hh[((size_t)l * 4 * Hh + n) * Hh + (k - Hh)];
  Wl[idx] = (bf16)v;
}

__global__ void pack_bias_kernel(float* __restrict__ bc, const float* __restrict__ b_ih,
                                 const float* __restrict__ b_hh) {
  const int idx = blockIdx.x * blockDim.x + threadIdx.x;      // L*4H
  if (idx >= Ll * 4 * Hh) return;
  bc[idx] = b_ih[idx] + b_hh[idx];
}

__global__ void conv_WaT_kernel(bf16* __restrict__ out, const float* __restrict__ Wa) {
  const int idx = blockIdx.x * blockDim.x + threadIdx.x;      // H*H, out[n=h][k=d]
  if (idx >= Hh * Hh) return;
  const int n = idx >> 9, d = idx & 511;
  out[idx] = (bf16)Wa[(size_t)d * Hh + n];
}

__global__ void conv_f32_bf16_kernel(bf16* __restrict__ out, const float* __restrict__ in,
                                     int count) {
  const int idx = blockIdx.x * blockDim.x + threadIdx.x;
  if (idx >= count) return;
  out[idx] = (bf16)in[idx];
}

__global__ void init_state_kernel(bf16* __restrict__ hbf, float* __restrict__ c) {
  const int idx = blockIdx.x * blockDim.x + threadIdx.x;      // L*B*H
  if (idx >= Ll * Bb * Hh) return;
  hbf[idx] = (bf16)0.f;
  c[idx] = 0.f;
}

__global__ void embed_kernel(bf16* __restrict__ xbf, const bf16* __restrict__ embbf,
                             const int* __restrict__ tok, const int* __restrict__ sot) {
  const int idx = blockIdx.x * blockDim.x + threadIdx.x;      // B*H
  if (idx >= Bb * Hh) return;
  const int b = idx >> 9, e = idx & 511;
  const int t = tok ? tok[b] : sot[0];
  xbf[idx] = embbf[(size_t)t * Hh + e];
}

__global__ void lstm_cell_kernel(const float* __restrict__ gates, float* __restrict__ c,
                                 bf16* __restrict__ hbf) {
  const int idx = blockIdx.x * blockDim.x + threadIdx.x;      // B*H
  if (idx >= Bb * Hh) return;
  const int b = idx >> 9, h = idx & 511;
  const float* g = gates + (size_t)b * (4 * Hh);
  const float ig = g[h], fg = g[Hh + h], gg = g[2 * Hh + h], og = g[3 * Hh + h];
  const float cn = sigm(fg) * c[idx] + sigm(ig) * tanhf(gg);
  c[idx] = cn;
  hbf[idx] = (bf16)(sigm(og) * tanhf(cn));
}

// one block per batch row: scores -> masked softmax -> ctx / ctx_emb
__global__ __launch_bounds__(256) void attn_kernel(
    const bf16* __restrict__ Whs,            // [S,B,H] bf16
    const bf16* __restrict__ hid,            // [B,H]   bf16 (top LSTM layer)
    const unsigned char* __restrict__ mask,  // [B,S]   bool
    const bf16* __restrict__ srco,           // [S,B,H] bf16
    const float* __restrict__ srce,          // [S,B,H] f32
    float* __restrict__ almt_out,            // [B,S]   (slice of d_out)
    bf16* __restrict__ ctx_bf,               // [B,H]
    float* __restrict__ ctx_emb) {           // [B,H]
  const int b = blockIdx.x;
  const int tid = threadIdx.x, lane = tid & 31, wv = tid >> 5;
  __shared__ bf16 sh_hid[Hh];
  __shared__ float sc[Ss];
  sh_hid[tid]       = hid[(size_t)b * Hh + tid];
  sh_hid[tid + 256] = hid[(size_t)b * Hh + tid + 256];
  __syncthreads();

  for (int s = wv; s < Ss; s += 8) {
    const bf16* wr = Whs + ((size_t)s * Bb + b) * Hh;
    float a = 0.f;
    for (int h = lane; h < Hh; h += 32) a += (float)wr[h] * (float)sh_hid[h];
    a = wred_sum(a);
    if (lane == 0) sc[s] = mask[(size_t)b * Ss + s] ? a : NEG_FILL;
  }
  __syncthreads();

  if (wv == 0) {  // softmax over 64 scores in one wave
    float a0 = sc[lane], a1 = sc[lane + 32];
    float m = wred_max(fmaxf(a0, a1));
    const float e0 = __expf(a0 - m), e1 = __expf(a1 - m);
    const float inv = 1.f / wred_sum(e0 + e1);
    sc[lane] = e0 * inv;
    sc[lane + 32] = e1 * inv;
  }
  __syncthreads();
  if (tid < Ss) almt_out[(size_t)b * Ss + tid] = sc[tid];

#pragma unroll
  for (int rep = 0; rep < 2; ++rep) {
    const int h = tid + rep * 256;
    float co = 0.f, ce = 0.f;
    for (int s = 0; s < Ss; ++s) {
      const float w = sc[s];
      const size_t idx = ((size_t)s * Bb + b) * Hh + h;
      co += w * (float)srco[idx];
      ce += w * srce[idx];
    }
    ctx_bf[(size_t)b * Hh + h] = (bf16)co;
    ctx_emb[(size_t)b * Hh + h] = ce;
  }
}

// NormControlledResidual (relative): out = x1 + x2 * (0.2 * n1 / n2)
__global__ __launch_bounds__(256) void residual_kernel(
    const float* __restrict__ ctx_emb, const float* __restrict__ hid_cat,
    bf16* __restrict__ hidres) {
  const int b = blockIdx.x, tid = threadIdx.x, lane = tid & 31, wv = tid >> 5;
  const float* x1 = ctx_emb + (size_t)b * Hh;
  const float* x2 = hid_cat + (size_t)b * Hh;
  float s1 = 0.f, s2 = 0.f;
  for (int h = tid; h < Hh; h += 256) {
    const float a = x1[h], c = x2[h];
    s1 += a * a; s2 += c * c;
  }
  s1 = wred_sum(s1); s2 = wred_sum(s2);
  __shared__ float r1[8], r2[8];
  __shared__ float scale_s;
  if (lane == 0) { r1[wv] = s1; r2[wv] = s2; }
  __syncthreads();
  if (tid == 0) {
    float t1 = 0.f, t2 = 0.f;
    for (int i = 0; i < 8; ++i) { t1 += r1[i]; t2 += r2[i]; }
    const float n1 = sqrtf(t1) + 1e-8f, n2 = sqrtf(t2) + 1e-8f;
    scale_s = 0.2f * n1 / n2;
  }
  __syncthreads();
  const float sc = scale_s;
  for (int h = tid; h < Hh; h += 256)
    hidres[(size_t)b * Hh + h] = (bf16)(x1[h] + x2[h] * sc);
}

__global__ __launch_bounds__(256) void logsoftmax_kernel(
    const float* __restrict__ logits /*[B,Vp]*/, float* __restrict__ outp /*[B,V]*/) {
  const int b = blockIdx.x, tid = threadIdx.x, lane = tid & 31, wv = tid >> 5;
  const float* row = logits + (size_t)b * Vp;
  __shared__ float red[8];
  __shared__ float bcast;
  float m = -3.4e38f;
  for (int v = tid; v < Vv; v += 256) m = fmaxf(m, row[v]);
  m = wred_max(m);
  if (lane == 0) red[wv] = m;
  __syncthreads();
  if (tid == 0) {
    float t = red[0];
    for (int i = 1; i < 8; ++i) t = fmaxf(t, red[i]);
    bcast = t;
  }
  __syncthreads();
  const float mm = bcast;
  float s = 0.f;
  for (int v = tid; v < Vv; v += 256) s += __expf(row[v] - mm);
  s = wred_sum(s);
  __syncthreads();
  if (lane == 0) red[wv] = s;
  __syncthreads();
  if (tid == 0) {
    float t = 0.f;
    for (int i = 0; i < 8; ++i) t += red[i];
    bcast = logf(t);
  }
  __syncthreads();
  const float ls = bcast;
  float* orow = outp + (size_t)b * Vv;
  for (int v = tid; v < Vv; v += 256) orow[v] = row[v] - mm - ls;
}

// ---------------------------------------------------------------------------

static inline int ceil_div(long long a, int b) { return (int)((a + b - 1) / b); }

extern "C" void kernel_launch(void* const* d_in, const int* in_sizes, int n_in,
                              void* d_out, int out_size, void* d_ws, size_t ws_size,
                              hipStream_t stream) {
  (void)in_sizes; (void)n_in; (void)out_size; (void)ws_size;

  const int*   sot      = (const int*)d_in[0];
  const float* src_emb  = (const float*)d_in[1];   // [S,B,H] f32
  const float* src_out  = (const float*)d_in[2];   // [S,B,H] f32
  const unsigned char* mask = (const unsigned char*)d_in[3];  // [B,S] bool
  const int*   target   = (const int*)d_in[4];     // [T,B]
  const float* emb      = (const float*)d_in[5];   // [V,H]
  const float* w_ih     = (const float*)d_in[6];   // [L,4H,H]
  const float* w_hh     = (const float*)d_in[7];   // [L,4H,H]
  const float* b_ih     = (const float*)d_in[8];   // [L,4H]
  const float* b_hh     = (const float*)d_in[9];   // [L,4H]
  const float* Wa       = (const float*)d_in[10];  // [H,H]
  const float* W_hid    = (const float*)d_in[11];  // [H,2H]
  const float* b_hid    = (const float*)d_in[12];  // [H]
  float* out = (float*)d_out;                      // [T,B,V] ++ [T,B,S]

  // bump allocator on workspace (256B aligned)
  char* ws = (char*)d_ws;
  size_t off = 0;
  auto alloc = [&](size_t bytes) -> char* {
    char* p = ws + off;
    off += (bytes + 255) & ~(size_t)255;
    return p;
  };

  bf16*  emb_bf   = (bf16*)alloc((size_t)Vp * Hh * 2);            // padded vocab
  bf16*  Wl_bf    = (bf16*)alloc((size_t)Ll * 4 * Hh * 2 * Hh * 2);
  float* bias_c   = (float*)alloc((size_t)Ll * 4 * Hh * 4);
  bf16*  WaT_bf   = (bf16*)alloc((size_t)Hh * Hh * 2);
  bf16*  Whid_bf  = (bf16*)alloc((size_t)Hh * 2 * Hh * 2);
  bf16*  srco_bf  = (bf16*)alloc((size_t)Ss * Bb * Hh * 2);
  bf16*  Whs_bf   = (bf16*)alloc((size_t)Ss * Bb * Hh * 2);
  bf16*  x_bf     = (bf16*)alloc((size_t)Bb * Hh * 2);
  bf16*  h_bf     = (bf16*)alloc((size_t)Ll * Bb * Hh * 2);
  float* c_f      = (float*)alloc((size_t)Ll * Bb * Hh * 4);
  float* gates    = (float*)alloc((size_t)Bb * 4 * Hh * 4);
  bf16*  ctx_bf   = (bf16*)alloc((size_t)Bb * Hh * 2);
  float* ctx_emb  = (float*)alloc((size_t)Bb * Hh * 4);
  float* hid_cat  = (float*)alloc((size_t)Bb * Hh * 4);
  bf16*  hidres   = (bf16*)alloc((size_t)Bb * Hh * 2);
  float* logits   = (float*)alloc((size_t)Bb * Vp * 4);

  bf16* h0 = h_bf;
  bf16* h1 = h_bf + (size_t)Bb * Hh;
  float* c0 = c_f;
  float* c1 = c_f + (size_t)Bb * Hh;
  float* almt_base = out + (size_t)Tt * Bb * Vv;

  // ---- one-time setup (recomputed every call for determinism) ----
  conv_emb_kernel<<<ceil_div((long long)Vp * Hh, 256), 256, 0, stream>>>(emb_bf, emb);
  pack_w_kernel<<<ceil_div((long long)Ll * 4 * Hh * 2 * Hh, 256), 256, 0, stream>>>(
      Wl_bf, w_ih, w_hh);
  pack_bias_kernel<<<ceil_div(Ll * 4 * Hh, 256), 256, 0, stream>>>(bias_c, b_ih, b_hh);
  conv_WaT_kernel<<<ceil_div(Hh * Hh, 256), 256, 0, stream>>>(WaT_bf, Wa);
  conv_f32_bf16_kernel<<<ceil_div(Hh * 2 * Hh, 256), 256, 0, stream>>>(
      Whid_bf, W_hid, Hh * 2 * Hh);
  conv_f32_bf16_kernel<<<ceil_div((long long)Ss * Bb * Hh, 256), 256, 0, stream>>>(
      srco_bf, src_out, Ss * Bb * Hh);
  init_state_kernel<<<ceil_div(Ll * Bb * Hh, 256), 256, 0, stream>>>(h_bf, c_f);

  // Wh_s = src_outputs @ Wa   -> bf16 [S*B, H]
  gemm_nt<true><<<dim3(Hh / 128, (Ss * Bb) / 128), 256, 0, stream>>>(
      srco_bf, nullptr, WaT_bf, nullptr, Whs_bf, Ss * Bb, Hh, Hh, Hh);

  // ---- sequential decode ----
  for (int t = 0; t < Tt; ++t) {
    const int* tok = (t == 0) ? nullptr : (target + (size_t)(t - 1) * Bb);
    embed_kernel<<<ceil_div(Bb * Hh, 256), 256, 0, stream>>>(x_bf, emb_bf, tok, sot);

    // layer 0: gates = [x|h0] @ W0^T + b0
    gemm_nt<false><<<dim3((4 * Hh) / 128, Bb / 128), 256, 0, stream>>>(
        x_bf, h0, Wl_bf, bias_c, gates, Bb, 4 * Hh, 2 * Hh, Hh);
    lstm_cell_kernel<<<ceil_div(Bb * Hh, 256), 256, 0, stream>>>(gates, c0, h0);

    // layer 1: gates = [h0_new|h1_prev] @ W1^T + b1
    gemm_nt<false><<<dim3((4 * Hh) / 128, Bb / 128), 256, 0, stream>>>(
        h0, h1, Wl_bf + (size_t)4 * Hh * 2 * Hh, bias_c + 4 * Hh,
        gates, Bb, 4 * Hh, 2 * Hh, Hh);
    lstm_cell_kernel<<<ceil_div(Bb * Hh, 256), 256, 0, stream>>>(gates, c1, h1);

    // attention: scores/softmax/ctx/ctx_emb; writes almts[t] directly
    attn_kernel<<<Bb, 256, 0, stream>>>(Whs_bf, h1, mask, srco_bf, src_emb,
                                        almt_base + (size_t)t * Bb * Ss,
                                        ctx_bf, ctx_emb);

    // hid_cat = [hid|ctx] @ W_hid^T + b_hid
    gemm_nt<false><<<dim3(Hh / 128, Bb / 128), 256, 0, stream>>>(
        h1, ctx_bf, Whid_bf, b_hid, hid_cat, Bb, Hh, 2 * Hh, Hh);

    residual_kernel<<<Bb, 256, 0, stream>>>(ctx_emb, hid_cat, hidres);

    // tied projection: logits = hid_res @ emb^T (vocab padded to 1024)
    gemm_nt<false><<<dim3(Vp / 128, Bb / 128), 256, 0, stream>>>(
        hidres, nullptr, emb_bf, nullptr, logits, Bb, Vp, Hh, Hh);

    logsoftmax_kernel<<<Bb, 256, 0, stream>>>(logits, out + (size_t)t * Bb * Vv);
  }
}
